// Attention_5909874999390
// MI455X (gfx1250) — compile-verified, matrix operand order checked by reading, not run on hardware
//
#include <hip/hip_runtime.h>

// Bahdanau attention, MI455X (gfx1250, wave32).
// Big GEMM (50176x512x2048, 105 GFLOP) done in bf16 WMMA with f32 accumulate;
// hidden[B,L,512] is never materialized (fused ReLU + W_att reduction).
// A-tile pipeline is double-buffered in LDS: one barrier per K-step, next
// tile's global loads overlap the current step's 16 WMMAs.

#define B_   256
#define L_   196
#define ENC_ 2048
#define DEC_ 512
#define ATT_ 512

typedef __attribute__((ext_vector_type(16))) __bf16          v16bf;
typedef __attribute__((ext_vector_type(8)))  __bf16          bf16x8;
typedef __attribute__((ext_vector_type(8)))  float           v8f;
typedef __attribute__((ext_vector_type(4)))  float           v4f;
typedef __attribute__((ext_vector_type(8)))  unsigned short  u16x8;

union FragAB { v16bf v; bf16x8 h[2]; };

__device__ __forceinline__ unsigned short f32_to_bf16_rne(float f) {
    unsigned int u = __float_as_uint(f);
    unsigned int r = u + 0x7FFFu + ((u >> 16) & 1u);
    return (unsigned short)(r >> 16);
}

__device__ __forceinline__ u16x8 cvt8_bf16(v4f f0, v4f f1) {
    u16x8 o;
    o[0] = f32_to_bf16_rne(f0[0]); o[1] = f32_to_bf16_rne(f0[1]);
    o[2] = f32_to_bf16_rne(f0[2]); o[3] = f32_to_bf16_rne(f0[3]);
    o[4] = f32_to_bf16_rne(f1[0]); o[5] = f32_to_bf16_rne(f1[1]);
    o[6] = f32_to_bf16_rne(f1[2]); o[7] = f32_to_bf16_rne(f1[3]);
    return o;
}

// ---------------------------------------------------------------------------
// Kernel 1: c[b,a] = decoder_hidden[b,:] @ W_dec[:,a] + b_enc[a] + b_dec[a]
// ---------------------------------------------------------------------------
__global__ __launch_bounds__(256)
void dec_proj_kernel(const float* __restrict__ dh, const float* __restrict__ Wd,
                     const float* __restrict__ be, const float* __restrict__ bd,
                     float* __restrict__ c) {
    int idx = blockIdx.x * 256 + threadIdx.x;   // B_*ATT_ total
    int b = idx >> 9;
    int a = idx & (ATT_ - 1);
    float acc = be[a] + bd[a];
    const float* dhb = dh + b * DEC_;
    for (int k = 0; k < DEC_; ++k)
        acc = fmaf(dhb[k], Wd[k * ATT_ + a], acc);
    c[idx] = acc;
}

// ---------------------------------------------------------------------------
// Kernel 2: W_enc [ENC][ATT] f32  ->  WT [ATT][ENC] bf16 (transposed, K-major)
// ---------------------------------------------------------------------------
__global__ __launch_bounds__(256)
void wenc_cvt_kernel(const float* __restrict__ We, unsigned short* __restrict__ WT) {
    int idx = blockIdx.x * 256 + threadIdx.x;   // ENC_*ATT_ total
    int k = idx >> 9;
    int n = idx & (ATT_ - 1);
    WT[n * ENC_ + k] = f32_to_bf16_rne(We[idx]);
}

// ---------------------------------------------------------------------------
// Kernel 3: fused GEMM + bias + ReLU + dot(W_att):
//   att[r] = sum_a relu( enc[r,:]@W_enc[:,a] + c[b(r),a] ) * W_att[a]
// 64 rows per workgroup (50176/64 = 784 WGs), 8 waves, each wave owns a
// 64-column slab of N. 16 v_wmma_f32_16x16x32_bf16 per wave per K-step.
// ---------------------------------------------------------------------------
__global__ __launch_bounds__(256)
void att_gemm_kernel(const float* __restrict__ enc,            // [B*L, ENC] f32
                     const unsigned short* __restrict__ WT,    // [ATT][ENC] bf16 bits
                     const float* __restrict__ c,              // [B, ATT]
                     const float* __restrict__ Wa,             // [ATT]
                     float* __restrict__ att)                  // [B*L]
{
    __shared__ unsigned short lA[2][64 * 32];  // double-buffered A tile, bf16
    __shared__ float          lP[64 * 128];    // per-(row, wave*16+lane) partials

    const int tid  = threadIdx.x;
    const int w    = tid >> 5;      // wave 0..7 -> N columns [64w, 64w+64)
    const int lane = tid & 31;
    const int half = lane >> 4;     // K-half selector (WMMA lane layout)
    const int lrow = lane & 15;     // row (A) / col (B) within 16-tile
    const int row0 = blockIdx.x * 64;

    v8f acc[4][4];                  // [m-tile][n-tile], 16x16 f32 each
    for (int mt = 0; mt < 4; ++mt)
        for (int nt = 0; nt < 4; ++nt)
            acc[mt][nt] = (v8f)0.0f;

    // A-staging role of this thread: 8 consecutive K elements of one row
    const int arow = tid >> 2;          // 0..63
    const int akk  = (tid & 3) * 8;     // 0,8,16,24
    const float* gA = enc + (row0 + arow) * (long)ENC_ + akk;

    // Prologue: stage K-step 0 into buffer 0
    {
        v4f f0 = *(const v4f*)gA;
        v4f f1 = *(const v4f*)(gA + 4);
        *(u16x8*)&lA[0][arow * 32 + akk] = cvt8_bf16(f0, f1);
    }
    __syncthreads();

    for (int it = 0; it < ENC_ / 32; ++it) {
        const int k0  = it * 32;
        const int cur = it & 1;
        const bool hasNext = (it < ENC_ / 32 - 1);

        // Prefetch next A tile into registers (overlaps the WMMAs below)
        v4f n0, n1;
        if (hasNext) {
            const float* g = gA + k0 + 32;
            n0 = *(const v4f*)g;
            n1 = *(const v4f*)(g + 4);
        }

        // B fragments straight from global (L2-resident, contiguous in K).
        // Lane layout (16-bit 16x32): lanes 0-15 hold K{0..7,16..23},
        // lanes 16-31 hold K{8..15,24..31}.
        FragAB bF[4];
        for (int nt = 0; nt < 4; ++nt) {
            const unsigned short* bp =
                WT + (w * 64 + nt * 16 + lrow) * ENC_ + k0 + half * 8;
            bF[nt].h[0] = *(const bf16x8*)bp;          // K 0..7   / 8..15
            bF[nt].h[1] = *(const bf16x8*)(bp + 16);   // K 16..23 / 24..31
        }
        FragAB aF[4];
        for (int mt = 0; mt < 4; ++mt) {
            const unsigned short* ap = &lA[cur][(mt * 16 + lrow) * 32 + half * 8];
            aF[mt].h[0] = *(const bf16x8*)ap;
            aF[mt].h[1] = *(const bf16x8*)(ap + 16);
        }
        for (int mt = 0; mt < 4; ++mt)
            for (int nt = 0; nt < 4; ++nt)
                acc[mt][nt] = __builtin_amdgcn_wmma_f32_16x16x32_bf16(
                    false, aF[mt].v, false, bF[nt].v,
                    (short)0, acc[mt][nt], false, false);

        // Stage next tile into the other buffer; one barrier per iteration is
        // enough with two buffers (protects RAW on next buf and WAR on cur).
        if (hasNext)
            *(u16x8*)&lA[cur ^ 1][arow * 32 + akk] = cvt8_bf16(n0, n1);
        __syncthreads();
    }

    // Epilogue: + c[b,a], ReLU, * W_att[a], partial N-reduction.
    float wa[4];
    for (int nt = 0; nt < 4; ++nt)
        wa[nt] = Wa[w * 64 + nt * 16 + lrow];

    for (int mt = 0; mt < 4; ++mt) {
        for (int j = 0; j < 8; ++j) {
            // C/D layout: VGPR j -> M=j (lanes 0-15) / M=j+8 (lanes 16-31), N=lane&15
            int rloc = mt * 16 + j + half * 8;
            int rg   = row0 + rloc;          // < 50176: 32-bit div is enough
            int bb   = rg / L_;
            const float* cb = c + bb * ATT_;
            float s = 0.0f;
            for (int nt = 0; nt < 4; ++nt) {
                int a = w * 64 + nt * 16 + lrow;
                float v = acc[mt][nt][j] + cb[a];
                v = fmaxf(v, 0.0f);
                s = fmaf(v, wa[nt], s);
            }
            lP[rloc * 128 + w * 16 + lrow] = s;
        }
    }
    __syncthreads();
    if (tid < 64) {
        float s = 0.0f;
        for (int i = 0; i < 128; ++i) s += lP[tid * 128 + i];
        att[row0 + tid] = s;   // b_att omitted: uniform shift, softmax-invariant
    }
}

// ---------------------------------------------------------------------------
// Kernel 4: alpha[b,:] = softmax(att[b,:]) over L
// ---------------------------------------------------------------------------
__global__ __launch_bounds__(256)
void softmax_kernel(const float* __restrict__ att, float* __restrict__ alpha) {
    __shared__ float sm[256];
    int b = blockIdx.x, t = threadIdx.x;
    float v = (t < L_) ? att[b * L_ + t] : -3.4e38f;
    sm[t] = v;
    __syncthreads();
    for (int s = 128; s > 0; s >>= 1) {
        if (t < s) sm[t] = fmaxf(sm[t], sm[t + s]);
        __syncthreads();
    }
    float mx = sm[0];
    __syncthreads();
    float e = (t < L_) ? __expf(v - mx) : 0.0f;
    sm[t] = e;
    __syncthreads();
    for (int s = 128; s > 0; s >>= 1) {
        if (t < s) sm[t] += sm[t + s];
        __syncthreads();
    }
    float inv = 1.0f / sm[0];
    if (t < L_) alpha[b * L_ + t] = e * inv;
}

// ---------------------------------------------------------------------------
// Kernel 5: awe[b,e] = sum_l enc[b,l,e] * alpha[b,l]   (pure-bandwidth)
// grid = B*4, each WG covers 512 e's (2 per thread), coalesced along e.
// ---------------------------------------------------------------------------
__global__ __launch_bounds__(256)
void awe_kernel(const float* __restrict__ enc, const float* __restrict__ alpha,
                float* __restrict__ awe) {
    __shared__ float al[L_];
    int b  = blockIdx.x >> 2;
    int e0 = (blockIdx.x & 3) * 512 + threadIdx.x;
    if (threadIdx.x < L_) al[threadIdx.x] = alpha[b * L_ + threadIdx.x];
    __syncthreads();
    float a0 = 0.0f, a1 = 0.0f;
    const float* p = enc + (long)b * L_ * ENC_ + e0;
    for (int l = 0; l < L_; ++l) {
        float wl = al[l];
        a0 = fmaf(p[0],   wl, a0);
        a1 = fmaf(p[256], wl, a1);
        p += ENC_;
    }
    awe[(long)b * ENC_ + e0]       = a0;
    awe[(long)b * ENC_ + e0 + 256] = a1;
}

// ---------------------------------------------------------------------------
extern "C" void kernel_launch(void* const* d_in, const int* in_sizes, int n_in,
                              void* d_out, int out_size, void* d_ws, size_t ws_size,
                              hipStream_t stream) {
    const float* enc = (const float*)d_in[0];
    const float* dh  = (const float*)d_in[1];
    const float* We  = (const float*)d_in[2];
    const float* be  = (const float*)d_in[3];
    const float* Wd  = (const float*)d_in[4];
    const float* bd  = (const float*)d_in[5];
    const float* Wa  = (const float*)d_in[6];
    // d_in[7] = b_att: uniform softmax shift, does not affect outputs.

    char* ws = (char*)d_ws;
    float*          c_tab = (float*)ws;                           // 256*512 f32
    float*          attw  = (float*)(ws + 524288);                // 256*196 f32
    unsigned short* WT    = (unsigned short*)(ws + 724992);       // 512*2048 bf16

    float* awe   = (float*)d_out;                 // [B, ENC]
    float* alpha = awe + (size_t)B_ * ENC_;       // [B, L]

    dec_proj_kernel<<<(B_ * ATT_) / 256, 256, 0, stream>>>(dh, Wd, be, bd, c_tab);
    wenc_cvt_kernel<<<(ENC_ * ATT_) / 256, 256, 0, stream>>>(We, WT);
    att_gemm_kernel<<<(B_ * L_) / 64, 256, 0, stream>>>(enc, WT, c_tab, Wa, attw);
    softmax_kernel<<<B_, 256, 0, stream>>>(attw, alpha);
    awe_kernel<<<B_ * 4, 256, 0, stream>>>(enc, alpha, awe);
}
